// NbeGNN_15650860826706
// MI455X (gfx1250) — matrix operand order repaired and checked by reading, not compiled, not run on hardware
//
#include <hip/hip_runtime.h>
#include <hip/hip_bf16.h>

#define NN   100000
#define EE   1600000
#define IND  128
#define HD   128
#define OUTD 64

typedef __attribute__((ext_vector_type(16))) __bf16 v16bf;
typedef __attribute__((ext_vector_type(8)))  float  v8f;

__device__ __forceinline__ __bf16 tobf(float f) { return (__bf16)f; }

// ---------------- elementwise helpers ----------------

__global__ void fill_kernel(float* __restrict__ p, float v, int n) {
  int i = blockIdx.x * blockDim.x + threadIdx.x;
  if (i < n) p[i] = v;
}

__global__ void degree_kernel(const long long* __restrict__ dst,
                              float* __restrict__ deg, int e) {
  int i = blockIdx.x * blockDim.x + threadIdx.x;
  if (i < e) atomicAdd(&deg[(int)dst[i]], 1.0f);
}

__global__ void rsqrt_kernel(float* __restrict__ d, int n) {
  int i = blockIdx.x * blockDim.x + threadIdx.x;
  if (i < n) d[i] = rsqrtf(d[i]);   // deg >= 1 from self-loop init
}

__global__ void bias_relu_kernel(float* __restrict__ h,
                                 const float* __restrict__ b, int n) {
  int i = blockIdx.x * blockDim.x + threadIdx.x;
  if (i < n) {
    float v = h[i] + b[i & (HD - 1)];
    h[i] = v > 0.0f ? v : 0.0f;
  }
}

__global__ void finalize_kernel(float* __restrict__ o,
                                const float* __restrict__ b, int n) {
  int i = blockIdx.x * blockDim.x + threadIdx.x;
  if (i < n) {
    float v = o[i] + b[i & (OUTD - 1)];
    o[i] = 0.8f * (1.0f / (1.0f + __expf(-v))) + 0.1f;
  }
}

// ---------------- WMMA GEMM: Y[M x Ncol] = X[M x 128] * W[128 x Ncol] ----------------
// One 16-row strip (all Ncol/16 col tiles) per wave, K=128 in four 16x16x32 bf16 WMMAs per col tile.
// W is converted to bf16 and transposed into LDS ([n][k]) so each lane's
// B-fragment (col n0+lane&15, K = kb + 16*half .. +15) is one contiguous 32B read.
__global__ void gemm_wmma_bf16(const float* __restrict__ X,
                               const float* __restrict__ W,
                               float* __restrict__ Y,
                               int M, int Ncol) {
  __shared__ __bf16 Wt[HD * HD];   // [n][k], up to 32 KB used
  const int K = 128;
  int tid = threadIdx.x;

  // cooperative convert + transpose of W (row-major [k][n]) into Wt[n][k]
  int total = K * Ncol;
  for (int idx = tid; idx < total; idx += blockDim.x) {
    int k = idx / Ncol;
    int n = idx - k * Ncol;
    Wt[n * K + k] = tobf(W[idx]);
  }
  __syncthreads();

  int wave = tid >> 5;
  int lane = tid & 31;
  int nl   = lane & 15;      // row-in-tile (A) / col-in-tile (B,D)
  int half = lane >> 4;      // K-half selector

  int rowTiles = M >> 4;
  int rt = blockIdx.x * (blockDim.x >> 5) + wave;   // wave-uniform guard: EXEC all-1s
  if (rt >= rowTiles) return;
  int m0 = rt << 4;

  // A fragments: lane holds row m0+nl; per 32-chunk, elem e -> K=kb+16*half+e
  // (contiguous K order, matching the B fragment)
  const float* xr = X + (size_t)(m0 + nl) * K;
  v16bf a[4];
#pragma unroll
  for (int c = 0; c < 4; ++c) {
    int kb = c << 5;
    const float4* g0 = (const float4*)(xr + kb + 16 * half);
    const float4* g1 = (const float4*)(xr + kb + 16 * half + 8);
    float4 f0 = g0[0], f1 = g0[1];
    float4 f2 = g1[0], f3 = g1[1];
    v16bf t;
    t[0]  = tobf(f0.x); t[1]  = tobf(f0.y); t[2]  = tobf(f0.z); t[3]  = tobf(f0.w);
    t[4]  = tobf(f1.x); t[5]  = tobf(f1.y); t[6]  = tobf(f1.z); t[7]  = tobf(f1.w);
    t[8]  = tobf(f2.x); t[9]  = tobf(f2.y); t[10] = tobf(f2.z); t[11] = tobf(f2.w);
    t[12] = tobf(f3.x); t[13] = tobf(f3.y); t[14] = tobf(f3.z); t[15] = tobf(f3.w);
    a[c] = t;
  }

  int colTiles = Ncol >> 4;
  for (int ct = 0; ct < colTiles; ++ct) {
    int n0 = ct << 4;
    v8f acc = {};
#pragma unroll
    for (int c = 0; c < 4; ++c) {
      int kb = c << 5;
      const v16bf* bp = (const v16bf*)&Wt[(n0 + nl) * K + kb + 16 * half];
      v16bf bfrag = *bp;
      acc = __builtin_amdgcn_wmma_f32_16x16x32_bf16(
          /*neg_a=*/false, a[c], /*neg_b=*/false, bfrag,
          /*c_mod=*/(short)0, acc, /*reuse_a=*/false, /*reuse_b=*/false);
    }
    // D layout: VGPR r -> row m0 + 8*half + r, col n0 + nl
    float* yr = Y + (size_t)(m0 + 8 * half) * Ncol + n0 + nl;
#pragma unroll
    for (int r = 0; r < 8; ++r)
      yr[(size_t)r * Ncol] = acc[r];
  }
}

// ---------------- edge aggregation: wave per edge ----------------
// out[dst] += feat[src] * dinv[src]*dinv[dst]; edges >= EE are self-loops.

__global__ void agg128_kernel(const long long* __restrict__ src,
                              const long long* __restrict__ dst,
                              const float* __restrict__ dinv,
                              const float* __restrict__ feat,
                              float* __restrict__ out, int etot) {
  int t = blockIdx.x * blockDim.x + threadIdx.x;
  int e = t >> 5;
  int lane = t & 31;
  if (e >= etot) return;
  int s, d;
  if (e < EE) { s = (int)src[e]; d = (int)dst[e]; }
  else        { s = e - EE; d = s; }
  float w = dinv[s] * dinv[d];
  float4 v = ((const float4*)(feat + (size_t)s * 128))[lane];
  float* o = out + (size_t)d * 128 + lane * 4;
  atomicAdd(o + 0, v.x * w);
  atomicAdd(o + 1, v.y * w);
  atomicAdd(o + 2, v.z * w);
  atomicAdd(o + 3, v.w * w);
}

__global__ void agg64_kernel(const long long* __restrict__ src,
                             const long long* __restrict__ dst,
                             const float* __restrict__ dinv,
                             const float* __restrict__ feat,
                             float* __restrict__ out, int etot) {
  int t = blockIdx.x * blockDim.x + threadIdx.x;
  int e = t >> 5;
  int lane = t & 31;
  if (e >= etot) return;
  int s, d;
  if (e < EE) { s = (int)src[e]; d = (int)dst[e]; }
  else        { s = e - EE; d = s; }
  float w = dinv[s] * dinv[d];
  float2 v = ((const float2*)(feat + (size_t)s * 64))[lane];
  float* o = out + (size_t)d * 64 + lane * 2;
  atomicAdd(o + 0, v.x * w);
  atomicAdd(o + 1, v.y * w);
}

// ---------------- driver ----------------

extern "C" void kernel_launch(void* const* d_in, const int* in_sizes, int n_in,
                              void* d_out, int out_size, void* d_ws, size_t ws_size,
                              hipStream_t stream) {
  const float*     x   = (const float*)d_in[0];
  const long long* ei  = (const long long*)d_in[1];   // (2, E) int64
  const float*     W1  = (const float*)d_in[2];
  const float*     b1  = (const float*)d_in[3];
  const float*     W2  = (const float*)d_in[4];
  const float*     b2  = (const float*)d_in[5];
  float*           out = (float*)d_out;

  const long long* src = ei;
  const long long* dst = ei + EE;

  char* ws = (char*)d_ws;
  float* deg  = (float*)ws;                                   // N floats; becomes dinv
  size_t off  = (((size_t)NN * 4) + 255) & ~(size_t)255;
  float* h1   = (float*)(ws + off);                           // N*128 f32 (reused as h2)
  float* agg1 = h1 + (size_t)NN * HD;                         // N*128 f32 (becomes h)

  const int B = 256;
  // init: deg=1 (self loop), agg1=0, out=0 (agg2 accumulates into d_out)
  fill_kernel<<<(NN + B - 1) / B, B, 0, stream>>>(deg, 1.0f, NN);
  fill_kernel<<<(NN * HD + B - 1) / B, B, 0, stream>>>(agg1, 0.0f, NN * HD);
  fill_kernel<<<(NN * OUTD + B - 1) / B, B, 0, stream>>>(out, 0.0f, NN * OUTD);

  degree_kernel<<<(EE + B - 1) / B, B, 0, stream>>>(dst, deg, EE);
  rsqrt_kernel<<<(NN + B - 1) / B, B, 0, stream>>>(deg, NN);

  const int rowTiles = NN / 16;                      // 6250, exact
  const int gemmBlocks = (rowTiles + 7) / 8;         // 8 waves / block

  // layer 1: h1 = x @ W1 ; aggregate ; +b1 ; relu (in agg1)
  gemm_wmma_bf16<<<gemmBlocks, B, 0, stream>>>(x, W1, h1, NN, HD);
  int etot = EE + NN;
  int aggBlocks = (etot * 32 + B - 1) / B;
  agg128_kernel<<<aggBlocks, B, 0, stream>>>(src, dst, deg, h1, agg1, etot);
  bias_relu_kernel<<<(NN * HD + B - 1) / B, B, 0, stream>>>(agg1, b1, NN * HD);

  // layer 2: h2 = h @ W2 (h2 reuses h1 buffer) ; aggregate into d_out ; sigmoid
  gemm_wmma_bf16<<<gemmBlocks, B, 0, stream>>>(agg1, W2, h1, NN, OUTD);
  agg64_kernel<<<aggBlocks, B, 0, stream>>>(src, dst, deg, h1, out, etot);
  finalize_kernel<<<(NN * OUTD + B - 1) / B, B, 0, stream>>>(out, b2, NN * OUTD);
}
